// SparseMultiHeadAttention_69303592288269
// MI455X (gfx1250) — matrix operand order repaired and checked by reading, not compiled, not run on hardware
//
#include <hip/hip_runtime.h>
#include <hip/hip_bf16.h>
#include <math.h>

#define S_LEN  2048
#define DMODEL 1024
#define NHEAD  16
#define HDIM   64
#define KTOP   204
#define BSEQ   4096   // B*S
#define BH     32     // B*H

typedef __attribute__((ext_vector_type(16))) __bf16 v16bf;
typedef __attribute__((ext_vector_type(8)))  float  v8f;

union FragU { v16bf v; unsigned u[8]; };

static __device__ __forceinline__ unsigned bf16_1(float f) {
  unsigned u = __float_as_uint(f);
  return (u + 0x7FFFu + ((u >> 16) & 1u)) >> 16;
}
static __device__ __forceinline__ unsigned bf16_2(float lo, float hi) {
  return bf16_1(lo) | (bf16_1(hi) << 16);
}
static __device__ __forceinline__ unsigned short bf16_s(float f) {
  return (unsigned short)bf16_1(f);
}

static __device__ __forceinline__ void ld16(unsigned* d, const unsigned short* p) {
  const uint4* q = reinterpret_cast<const uint4*>(p);
  uint4 t = *q;
  d[0] = t.x; d[1] = t.y; d[2] = t.z; d[3] = t.w;
}

// A fragment (16x32 bf16): element (m,k) at base + m*ld + k.
// ISA layout: lane<16 -> m=lane, K pairs {0..7, 16..23}; lane>=16 -> m=lane-16, K {8..15, 24..31}.
static __device__ __forceinline__ v16bf afrag(const unsigned short* base, int ld, int lane) {
  FragU f;
  const int m = lane & 15, ko = (lane >> 4) * 8;
  ld16(f.u,     base + m * ld + ko);
  ld16(f.u + 4, base + m * ld + 16 + ko);
  return f.v;
}
// B fragment (32x16 bf16): element (k,n) at base + n*ld + k.
// ISA layout: lane<16 -> n=lane, K {0..7, 16..23}; lane>=16 -> n=lane-16, K {8..15, 24..31}.
static __device__ __forceinline__ v16bf bfrag(const unsigned short* base, int ld, int lane) {
  FragU f;
  const int n = lane & 15, ko = (lane >> 4) * 8;
  ld16(f.u,     base + n * ld + ko);
  ld16(f.u + 4, base + n * ld + ko + 16);
  return f.v;
}
static __device__ __forceinline__ v8f wmma_bf16(v16bf a, v16bf b, v8f c) {
  return __builtin_amdgcn_wmma_f32_16x16x32_bf16(false, a, false, b, (short)0, c, false, false);
}

#define V8F_ZERO {0.f, 0.f, 0.f, 0.f, 0.f, 0.f, 0.f, 0.f}

// ---------------------------------------------------------------------------
// Kernel 1: Q/K/V projection. X[4096,1024]f32 @ W^T + b -> bf16.
// mode 0: Q (scaled by 1/8) -> [B,H,S,HD]; mode 1: K -> [B,H,S,HD];
// mode 2: V -> transposed [B,H,HD,S] for contiguous B-fragments in attn*V.
// ---------------------------------------------------------------------------
__global__ __launch_bounds__(256) void qkv_proj_kernel(
    const float* __restrict__ X,
    const float* __restrict__ Wq, const float* __restrict__ bq,
    const float* __restrict__ Wk, const float* __restrict__ bk,
    const float* __restrict__ Wv, const float* __restrict__ bv,
    unsigned short* __restrict__ Qbf, unsigned short* __restrict__ Kbf,
    unsigned short* __restrict__ Vt)
{
  const int mode = blockIdx.z;
  const float* W    = (mode == 0) ? Wq : (mode == 1) ? Wk : Wv;
  const float* bias = (mode == 0) ? bq : (mode == 1) ? bk : bv;

  __shared__ unsigned short Xs[64 * 40];
  __shared__ unsigned short Ws[64 * 40];

  const int t = threadIdx.x, wave = t >> 5, lane = t & 31;
  const int row0 = blockIdx.x * 64;   // sequence rows
  const int col0 = blockIdx.y * 64;   // output features
  const int lr = t >> 2, lc = (t & 3) * 8;

  v8f acc0 = V8F_ZERO, acc1 = V8F_ZERO;
  const int rt = wave & 3, ct = (wave >> 2) * 2;

  for (int kb = 0; kb < DMODEL; kb += 32) {
    {
      const float* s = X + (size_t)(row0 + lr) * DMODEL + kb + lc;
      float4 a = *(const float4*)s, b = *(const float4*)(s + 4);
      unsigned* d = (unsigned*)&Xs[lr * 40 + lc];
      d[0] = bf16_2(a.x, a.y); d[1] = bf16_2(a.z, a.w);
      d[2] = bf16_2(b.x, b.y); d[3] = bf16_2(b.z, b.w);
    }
    {
      const float* s = W + (size_t)(col0 + lr) * DMODEL + kb + lc;
      float4 a = *(const float4*)s, b = *(const float4*)(s + 4);
      unsigned* d = (unsigned*)&Ws[lr * 40 + lc];
      d[0] = bf16_2(a.x, a.y); d[1] = bf16_2(a.z, a.w);
      d[2] = bf16_2(b.x, b.y); d[3] = bf16_2(b.z, b.w);
    }
    __syncthreads();
    v16bf a  = afrag(&Xs[rt * 16 * 40], 40, lane);
    v16bf b0 = bfrag(&Ws[(ct * 16) * 40], 40, lane);
    v16bf b1 = bfrag(&Ws[((ct + 1) * 16) * 40], 40, lane);
    acc0 = wmma_bf16(a, b0, acc0);
    acc1 = wmma_bf16(a, b1, acc1);
    __syncthreads();
  }

  const int half = lane >> 4, n = lane & 15;
  const int bidx = row0 >> 11;   // batch index (block never crosses batch boundary)
  #pragma unroll
  for (int j = 0; j < 2; ++j) {
    v8f acc = j ? acc1 : acc0;
    const int colg = col0 + (ct + j) * 16 + n;
    const float bb = bias[colg];
    const int h = colg >> 6, hd = colg & 63;
    if (mode < 2) {
      unsigned short* dst = mode ? Kbf : Qbf;
      const float scale = (mode == 0) ? 0.125f : 1.0f;   // fold 1/sqrt(HD) into Q
      #pragma unroll
      for (int r = 0; r < 8; ++r) {
        const int sg = row0 + rt * 16 + r + 8 * half;
        const int ss = sg & (S_LEN - 1);
        const float v = (acc[r] + bb) * scale;
        dst[((size_t)(bidx * NHEAD + h) * S_LEN + ss) * HDIM + hd] = bf16_s(v);
      }
    } else {
      // V transposed: [B,H,HD,S]; lane's 8 acc values are 8 consecutive seq positions
      const int sbase = (row0 + rt * 16 + 8 * half) & (S_LEN - 1);
      float v[8];
      #pragma unroll
      for (int r = 0; r < 8; ++r) v[r] = acc[r] + bb;
      uint4 o;
      o.x = bf16_2(v[0], v[1]); o.y = bf16_2(v[2], v[3]);
      o.z = bf16_2(v[4], v[5]); o.w = bf16_2(v[6], v[7]);
      *(uint4*)&Vt[((size_t)(bidx * NHEAD + h) * HDIM + hd) * S_LEN + sbase] = o;
    }
  }
}

// ---------------------------------------------------------------------------
// Kernel 2: fused attention core per (b,h, 16-row q tile).
// scores (WMMA) -> LDS; exact top-204 threshold via radix select; softmax;
// single coalesced attn write (simultaneously packing attn to bf16 in LDS);
// ctx = attn @ V via WMMA with pure ds_load_b128 fragment feeds.
// ---------------------------------------------------------------------------
__global__ __launch_bounds__(256) void sparse_attn_kernel(
    const unsigned short* __restrict__ Qbf,
    const unsigned short* __restrict__ Kbf,
    const unsigned short* __restrict__ Vt,
    float* __restrict__ attn,
    unsigned short* __restrict__ ctx)
{
  __shared__ float sc[16 * 2052];   // 16 q-rows x 2048 scores (+pad)
  __shared__ int   hist[8 * 256];   // per-wave radix histograms
  __shared__ float pc[4 * 256];     // partial ctx exchange
  __shared__ float rscale[16];

  const int t = threadIdx.x, wave = t >> 5, lane = t & 31;
  const int bh = blockIdx.y, q0 = blockIdx.x * 16;
  const int half = lane >> 4, nl = lane & 15;

  // ---- phase 1: scores = (Q/8) @ K^T ----
  const unsigned short* Qb = Qbf + ((size_t)bh * S_LEN + q0) * HDIM;
  const v16bf a_lo = afrag(Qb, HDIM, lane);        // hd 0..31
  const v16bf a_hi = afrag(Qb + 32, HDIM, lane);   // hd 32..63
  for (int kt = wave; kt < S_LEN / 16; kt += 8) {
    const unsigned short* Kb = Kbf + ((size_t)bh * S_LEN + kt * 16) * HDIM;
    v16bf b_lo = bfrag(Kb, HDIM, lane);
    v16bf b_hi = bfrag(Kb + 32, HDIM, lane);
    v8f c = V8F_ZERO;
    c = wmma_bf16(a_lo, b_lo, c);
    c = wmma_bf16(a_hi, b_hi, c);
    #pragma unroll
    for (int r = 0; r < 8; ++r)
      sc[(r + 8 * half) * 2052 + kt * 16 + nl] = c[r];
  }
  __syncthreads();

  // ---- phase 2: per-row max, exact top-KTOP threshold, exp ----
  for (int rr = 0; rr < 2; ++rr) {
    const int row = wave * 2 + rr;
    float* rp = &sc[row * 2052];

    float mx = -INFINITY;
    for (int j = 0; j < 64; ++j) mx = fmaxf(mx, rp[lane + 32 * j]);
    #pragma unroll
    for (int off = 16; off > 0; off >>= 1) mx = fmaxf(mx, __shfl_xor(mx, off, 32));

    int* h = &hist[wave * 256];
    unsigned prefix = 0;
    int remaining = KTOP;
    for (int pass = 0; pass < 4; ++pass) {
      const int sh = 24 - pass * 8;
      const unsigned maskhi = (pass == 0) ? 0u : (0xFFFFFFFFu << (sh + 8));
      #pragma unroll
      for (int i = 0; i < 8; ++i) h[lane * 8 + i] = 0;
      for (int j = 0; j < 64; ++j) {
        unsigned u = __float_as_uint(rp[lane + 32 * j]);
        u ^= (u >> 31) ? 0xFFFFFFFFu : 0x80000000u;   // monotone key
        if ((u & maskhi) == prefix) atomicAdd(&h[(u >> sh) & 255], 1);
      }
      int gs = 0;
      #pragma unroll
      for (int i = 0; i < 8; ++i) gs += h[lane * 8 + i];
      int acc = 0, grp = -1;
      for (int g = 31; g >= 0; --g) {          // suffix scan over 8-bin groups
        int s0 = __shfl(gs, g, 32);
        if (grp < 0) {
          if (acc + s0 >= remaining) grp = g;
          else acc += s0;
        }
      }
      if (grp < 0) grp = 0;
      int sel = -1;
      for (int b2 = grp * 8 + 7; b2 >= grp * 8; --b2) {
        int c0 = h[b2];
        if (sel < 0) {
          if (acc + c0 >= remaining) sel = b2;
          else acc += c0;
        }
      }
      if (sel < 0) sel = grp * 8;
      prefix |= ((unsigned)sel) << sh;
      remaining -= acc;
    }
    unsigned tu = prefix;
    tu = (tu & 0x80000000u) ? (tu ^ 0x80000000u) : ~tu;
    const float thresh = __uint_as_float(tu);   // exact 204th-largest score

    float sum = 0.f;
    for (int j = 0; j < 64; ++j) {
      const float v = rp[lane + 32 * j];
      const float e = (v >= thresh) ? __expf(v - mx) : 0.f;
      rp[lane + 32 * j] = e;
      sum += e;
    }
    #pragma unroll
    for (int off = 16; off > 0; off >>= 1) sum += __shfl_xor(sum, off, 32);
    if (lane == 0) rscale[row] = 1.f / sum;
  }
  __syncthreads();

  // ---- phase 3: normalize, write attn (f32, coalesced), and pack the
  //      normalized row to bf16 IN PLACE in LDS (8 bf16 into the first 16 B
  //      of each thread's own 8-float chunk; the f32 copy is dead after this).
  float* arow = attn + ((size_t)bh * S_LEN + q0) * S_LEN;
  for (int r = 0; r < 16; ++r) {
    const float s = rscale[r];
    float* rp = &sc[r * 2052 + t * 8];
    float4 o0 = *(float4*)rp;
    float4 o1 = *(float4*)(rp + 4);
    o0.x *= s; o0.y *= s; o0.z *= s; o0.w *= s;
    o1.x *= s; o1.y *= s; o1.z *= s; o1.w *= s;
    *(float4*)(arow + (size_t)r * S_LEN + t * 8)     = o0;
    *(float4*)(arow + (size_t)r * S_LEN + t * 8 + 4) = o1;
    uint4 pk;
    pk.x = bf16_2(o0.x, o0.y); pk.y = bf16_2(o0.z, o0.w);
    pk.z = bf16_2(o1.x, o1.y); pk.w = bf16_2(o1.z, o1.w);
    *(uint4*)rp = pk;   // packed bf16 keys [t*8, t*8+8) live at chunk start
  }
  __syncthreads();

  // ---- phase 4: ctx = attn @ V (K split across wave halves) ----
  // Packed-chunk addressing: bf16 key k of row m lives in the 16 B at
  // float slot sc[m*2052 + 8*(k>>3)]; every fragment half spans one chunk.
  const int ntile = wave & 3, khalf = wave >> 2;
  const unsigned short* Vb = Vt + ((size_t)bh * HDIM + ntile * 16) * S_LEN + khalf * 1024;
  v8f acc = V8F_ZERO;
  const float* arow2 = &sc[nl * 2052 + (size_t)khalf * 1024 / 8 * 8];  // + khalf*1024 floats? no:
  // NOTE: packed chunks are indexed by key: chunk c holds keys [8c, 8c+8).
  // Key base for this wave is khalf*1024 -> chunk base khalf*128.
  const float* rowbase = &sc[nl * 2052];
  const int cbase = khalf * 128;
  for (int kk = 0; kk < 32; ++kk) {
    const int c0 = cbase + kk * 4 + half;       // keys kbase + {0..7 | 8..15}
    FragU fa;
    ld16(fa.u,     (const unsigned short*)(rowbase + 8 * c0));
    ld16(fa.u + 4, (const unsigned short*)(rowbase + 8 * (c0 + 2)));  // keys +16
    FragU fb;
    const unsigned short* vc = Vb + nl * S_LEN + kk * 32 + half * 8;
    ld16(fb.u, vc);
    ld16(fb.u + 4, vc + 16);
    acc = wmma_bf16(fa.v, fb.v, acc);
  }
  if (wave >= 4) {
    float* p = &pc[ntile * 256 + lane * 8];
    #pragma unroll
    for (int r = 0; r < 8; ++r) p[r] = acc[r];
  }
  __syncthreads();
  if (wave < 4) {
    const float* p = &pc[ntile * 256 + lane * 8];
    const int b = bh >> 4, hh = bh & 15;
    #pragma unroll
    for (int r = 0; r < 8; ++r) {
      const float v = acc[r] + p[r];
      const int m = r + 8 * half;
      ctx[(size_t)(b * S_LEN + q0 + m) * DMODEL + hh * 64 + ntile * 16 + nl] = bf16_s(v);
    }
  }
  (void)arow2;
}

// ---------------------------------------------------------------------------
// Kernel 3: out = ctx @ Wo^T + bo  (f32 result straight into d_out).
// ---------------------------------------------------------------------------
__global__ __launch_bounds__(256) void out_proj_kernel(
    const unsigned short* __restrict__ Abf,
    const float* __restrict__ W, const float* __restrict__ bias,
    float* __restrict__ Y)
{
  __shared__ unsigned short As[64 * 40];
  __shared__ unsigned short Ws[64 * 40];

  const int t = threadIdx.x, wave = t >> 5, lane = t & 31;
  const int row0 = blockIdx.x * 64, col0 = blockIdx.y * 64;
  const int lr = t >> 2, lc = (t & 3) * 8;
  const int rt = wave & 3, ct = (wave >> 2) * 2;

  v8f acc0 = V8F_ZERO, acc1 = V8F_ZERO;
  for (int kb = 0; kb < DMODEL; kb += 32) {
    *(uint4*)&As[lr * 40 + lc] =
        *(const uint4*)(Abf + (size_t)(row0 + lr) * DMODEL + kb + lc);
    {
      const float* s = W + (size_t)(col0 + lr) * DMODEL + kb + lc;
      float4 a = *(const float4*)s, b = *(const float4*)(s + 4);
      unsigned* d = (unsigned*)&Ws[lr * 40 + lc];
      d[0] = bf16_2(a.x, a.y); d[1] = bf16_2(a.z, a.w);
      d[2] = bf16_2(b.x, b.y); d[3] = bf16_2(b.z, b.w);
    }
    __syncthreads();
    v16bf a  = afrag(&As[rt * 16 * 40], 40, lane);
    v16bf b0 = bfrag(&Ws[(ct * 16) * 40], 40, lane);
    v16bf b1 = bfrag(&Ws[((ct + 1) * 16) * 40], 40, lane);
    acc0 = wmma_bf16(a, b0, acc0);
    acc1 = wmma_bf16(a, b1, acc1);
    __syncthreads();
  }

  const int half = lane >> 4, n = lane & 15;
  #pragma unroll
  for (int j = 0; j < 2; ++j) {
    v8f acc = j ? acc1 : acc0;
    const int colg = col0 + (ct + j) * 16 + n;
    const float bb = bias[colg];
    #pragma unroll
    for (int r = 0; r < 8; ++r) {
      const int rowg = row0 + rt * 16 + r + 8 * half;
      Y[(size_t)rowg * DMODEL + colg] = acc[r] + bb;
    }
  }
}

// ---------------------------------------------------------------------------
// Kernel 4: in-place residual + LayerNorm over each row of 1024.
// ---------------------------------------------------------------------------
__global__ __launch_bounds__(256) void layernorm_kernel(
    float* __restrict__ Y, const float* __restrict__ X,
    const float* __restrict__ gw, const float* __restrict__ bw)
{
  const int row = blockIdx.x, t = threadIdx.x;
  const size_t base = (size_t)row * DMODEL + t * 4;
  float4 y = *(const float4*)(Y + base);
  float4 x = *(const float4*)(X + base);
  const float v0 = y.x + x.x, v1 = y.y + x.y, v2 = y.z + x.z, v3 = y.w + x.w;
  float s1 = v0 + v1 + v2 + v3;
  float s2 = v0 * v0 + v1 * v1 + v2 * v2 + v3 * v3;
  #pragma unroll
  for (int off = 16; off > 0; off >>= 1) {
    s1 += __shfl_xor(s1, off, 32);
    s2 += __shfl_xor(s2, off, 32);
  }
  __shared__ float w1[8], w2[8];
  if ((t & 31) == 0) { w1[t >> 5] = s1; w2[t >> 5] = s2; }
  __syncthreads();
  float ts1 = 0.f, ts2 = 0.f;
  #pragma unroll
  for (int i = 0; i < 8; ++i) { ts1 += w1[i]; ts2 += w2[i]; }
  const float mean = ts1 * (1.f / DMODEL);
  const float var  = ts2 * (1.f / DMODEL) - mean * mean;
  const float inv  = rsqrtf(var + 1e-5f);
  float4 g4 = *(const float4*)(gw + t * 4);
  float4 b4 = *(const float4*)(bw + t * 4);
  float4 o;
  o.x = (v0 - mean) * inv * g4.x + b4.x;
  o.y = (v1 - mean) * inv * g4.y + b4.y;
  o.z = (v2 - mean) * inv * g4.z + b4.z;
  o.w = (v3 - mean) * inv * g4.w + b4.w;
  *(float4*)(Y + base) = o;
}

// ---------------------------------------------------------------------------
extern "C" void kernel_launch(void* const* d_in, const int* in_sizes, int n_in,
                              void* d_out, int out_size, void* d_ws, size_t ws_size,
                              hipStream_t stream)
{
  (void)in_sizes; (void)n_in; (void)out_size; (void)ws_size;
  const float* query = (const float*)d_in[0];
  const float* Wq = (const float*)d_in[1];
  const float* bq = (const float*)d_in[2];
  const float* Wk = (const float*)d_in[3];
  const float* bk = (const float*)d_in[4];
  const float* Wv = (const float*)d_in[5];
  const float* bv = (const float*)d_in[6];
  const float* Wo = (const float*)d_in[7];
  const float* bo = (const float*)d_in[8];
  const float* g  = (const float*)d_in[9];
  const float* bl = (const float*)d_in[10];

  const size_t NE = (size_t)BSEQ * DMODEL;   // 4,194,304 elements
  unsigned short* qbf = (unsigned short*)d_ws;          //  8 MB
  unsigned short* kbf = qbf + NE;                       //  8 MB
  unsigned short* vt  = kbf + NE;                       //  8 MB (V transposed)
  unsigned short* ctx = vt + NE;                        //  8 MB

  float* out  = (float*)d_out;          // [B,S,D] f32
  float* attn = out + NE;               // [B,H,S,S] f32

  qkv_proj_kernel<<<dim3(BSEQ / 64, DMODEL / 64, 3), 256, 0, stream>>>(
      query, Wq, bq, Wk, bk, Wv, bv, qbf, kbf, vt);
  sparse_attn_kernel<<<dim3(S_LEN / 16, BH), 256, 0, stream>>>(
      qbf, kbf, vt, attn, ctx);
  out_proj_kernel<<<dim3(BSEQ / 64, DMODEL / 64), 256, 0, stream>>>(
      ctx, Wo, bo, out);
  layernorm_kernel<<<BSEQ, 256, 0, stream>>>(out, query, g, bl);
}
